// SHD_86088324481700
// MI455X (gfx1250) — matrix operand order repaired and checked by reading, not compiled, not run on hardware
//
#include <hip/hip_runtime.h>

typedef _Float16 half_t;
typedef __attribute__((ext_vector_type(16))) _Float16 v16h;
typedef __attribute__((ext_vector_type(8)))  _Float16 v8h;
typedef __attribute__((ext_vector_type(8)))  float    v8f;
typedef __attribute__((ext_vector_type(4)))  float    v4f;

#define B_SZ   128
#define T_SZ   500
#define TP     512   // T padded to power of two for the GEMM M dimension
#define NIN    700
#define NINP   704   // NIN padded to 22*32
#define NH     200
#define NOUT   20
#define NHP    224   // NH padded to 7*32 (7 k-chunks, 14 n-tiles)
#define NBT    8     // batch tiles (128/16)
#define MROWS  (B_SZ * T_SZ)             // 64000 rows of x
#define MTILES (B_SZ * TP / 16)          // 4096 padded M-tiles
#define NTILES 14
#define TILES  (MTILES * NTILES)         // 57344 = 7168 blocks * 8 waves
#define TSTRIDE ((long)NBT * NHP * 16)   // cur_ff floats per time step

// cur_ff layout: [t][btile][n][m]  (m = batch-in-tile, contiguous 16)
// -> scan kernel reads 8 contiguous floats per lane per step (2x b128).
#define CUR_IDX(t, btile, n, m) \
  (((((long)(t)) * NBT + (btile)) * NHP + (n)) * 16 + (m))

// ---------------------------------------------------------------------------
// Kernel 0: zero-padded f32 -> f16 repack (rows x cols -> rowsP x colsP).
// ---------------------------------------------------------------------------
__global__ __launch_bounds__(256)
void pack_f16(const float* __restrict__ src, half_t* __restrict__ dst,
              int rows, int cols, int rowsP, int colsP) {
  long total = (long)rowsP * colsP;
  long stride = (long)gridDim.x * blockDim.x;
  for (long i = (long)blockIdx.x * blockDim.x + threadIdx.x; i < total; i += stride) {
    int r = (int)(i / colsP);
    int c = (int)(i % colsP);
    float v = (r < rows && c < cols) ? src[(long)r * cols + c] : 0.0f;
    dst[i] = (half_t)v;
  }
}

// ---------------------------------------------------------------------------
// Kernel 1: cur1_ff[b,t,n] = sum_k x[b,t,k] * W1[n,k] + b1[n]
// M padded to b*512+t: batch/time split is shifts. Branch-free inner chunk =
// 4x b128 loads + 1 wmma. Epilogue: single scalar branch; fully-valid tiles
// (31/32) store with one base index + constant increments.
// ---------------------------------------------------------------------------
__global__ __launch_bounds__(256)
void snn_ff_gemm(const half_t* __restrict__ xh,   // [MROWS, NINP] f16
                 const half_t* __restrict__ W1h,  // [NHP,   NINP] f16 padded
                 const float*  __restrict__ b1,
                 float* __restrict__ cur_ff) {
  const int lane    = threadIdx.x & 31;
  const int wave    = threadIdx.x >> 5;
  const int tile    = blockIdx.x * 8 + wave;     // 7168 * 8 = 57344 exact
  const int mt      = tile / NTILES;
  const int nt      = tile % NTILES;
  const int half_hi = (lane >= 16) ? 1 : 0;
  const int mn      = lane & 15;
  const int ncol    = nt * 16 + mn;              // B row (output column n)

  // A row: padded M -> (batch, time); clamp time so loads stay in-bounds
  // (rows with tt >= 500 are computed but never stored).
  const int gg   = mt * 16 + mn;
  const int bb_a = gg >> 9;
  const int tt_a = gg & (TP - 1);
  const int arow_i = bb_a * T_SZ + ((tt_a < T_SZ) ? tt_a : (T_SZ - 1));

  const half_t* arow = xh  + (long)arow_i * NINP + 8 * half_hi; // A: K=kb+h+8*(h>=8)+8*hi
  const half_t* brow = W1h + (long)ncol   * NINP + 8 * half_hi; // B: K=kb+h+8*(h>=8)+8*hi

  v8f acc = {};
#pragma unroll 2
  for (int kb = 0; kb < NINP; kb += 32) {
    v8h  alo = *(const v8h*)(arow + kb);
    v8h  ahi = *(const v8h*)(arow + kb + 16);
    v16h a   = __builtin_shufflevector(alo, ahi, 0,1,2,3,4,5,6,7,
                                                 8,9,10,11,12,13,14,15);
    v8h  blo = *(const v8h*)(brow + kb);
    v8h  bhi = *(const v8h*)(brow + kb + 16);
    v16h b   = __builtin_shufflevector(blo, bhi, 0,1,2,3,4,5,6,7,
                                                 8,9,10,11,12,13,14,15);
    acc = __builtin_amdgcn_wmma_f32_16x16x32_f16(false, a, false, b,
                                                 (short)0, acc, false, false);
  }
  const float bias = (ncol < NH) ? b1[ncol] : 0.0f;   // pad columns stay 0

  // D rows: g = g0 + r; same batch for all r (no 512-crossing inside tile).
  const int g0  = mt * 16 + half_hi * 8;
  const int bb  = g0 >> 9;
  const int tt0 = g0 & (TP - 1);
  long idx = CUR_IDX(tt0, bb >> 4, ncol, bb & 15);

  if ((mt & 31) != 31) {                          // scalar: tile fully valid
#pragma unroll
    for (int r = 0; r < 8; ++r) {
      cur_ff[idx] = acc[r] + bias;
      idx += TSTRIDE;
    }
  } else {                                        // rare boundary tile
#pragma unroll
    for (int r = 0; r < 8; ++r) {
      if (tt0 + r < T_SZ) cur_ff[idx] = acc[r] + bias;
      idx += TSTRIDE;
    }
  }
}

// ---------------------------------------------------------------------------
// Kernel 2: recurrent scan. 8 blocks x 512 threads; each block owns 16 batch
// rows (recurrence independent per row -> no inter-block sync). Waves 0..13:
// layer-1 N-tiles with Wfb fragments pinned in VGPRs; waves 14..15: layer-2
// with W2 fragments. Spikes exchanged via LDS (exact in f16).
// ---------------------------------------------------------------------------
__global__ __launch_bounds__(512)
void snn_scan(const float*  __restrict__ cur_ff,  // [T][NBT][NHP][16]
              const half_t* __restrict__ Wfbh,    // [NHP, NHP] f16 padded
              const half_t* __restrict__ W2h,     // [32,  NHP] f16 padded
              const float* __restrict__ bfb,  const float* __restrict__ b2,
              const float* __restrict__ alpha1, const float* __restrict__ beta1,
              const float* __restrict__ thr1,
              const float* __restrict__ alpha2, const float* __restrict__ beta2,
              const float* __restrict__ thr2,
              float* __restrict__ out) {
  __shared__ __align__(16) half_t spkLDS[16 * NHP];
  __shared__ float smax[16 * 32];

  const int tid     = threadIdx.x;
  const int lane    = tid & 31;
  const int wave    = tid >> 5;                  // 0..15
  const int half_hi = (lane >= 16) ? 1 : 0;
  const int mn      = lane & 15;
  const int btile   = blockIdx.x;                // batch tile (16 rows)
  const int b0      = btile * 16;

  for (int i = tid; i < 16 * NHP; i += 512) spkLDS[i] = (half_t)0.0f;

  const bool is_l1 = (wave < 14);
  const int  jt    = is_l1 ? wave : (wave - 14);
  const int  ncol  = jt * 16 + mn;               // n (L1) or o (L2)

  // Pin B-fragments (Wfb or W2) in VGPRs for the whole 500-step scan.
  const half_t* wrow = (is_l1 ? Wfbh : W2h) + (long)ncol * NHP + 8 * half_hi;
  v16h bfrag[7];
#pragma unroll
  for (int c = 0; c < 7; ++c) {
    v8h wlo = *(const v8h*)(wrow + c * 32);
    v8h whi = *(const v8h*)(wrow + c * 32 + 16);
    bfrag[c] = __builtin_shufflevector(wlo, whi, 0,1,2,3,4,5,6,7,
                                       8,9,10,11,12,13,14,15);
  }

  // Per-column neuron parameters (each lane owns one column in C/D layout).
  float p_alpha, p_beta, p_thr, p_bias;
  if (is_l1) {
    bool ok = ncol < NH;
    p_alpha = ok ? alpha1[ncol] : 0.0f;
    p_beta  = ok ? beta1 [ncol] : 0.0f;
    p_thr   = ok ? thr1  [ncol] : 1.0f;
    p_bias  = ok ? bfb   [ncol] : 0.0f;
  } else {
    bool ok = ncol < NOUT;
    p_alpha = ok ? alpha2[ncol] : 0.0f;
    p_beta  = ok ? beta2 [ncol] : 0.0f;
    p_thr   = ok ? thr2  [ncol] : 1.0f;
    p_bias  = ok ? b2    [ncol] : 0.0f;
  }

  v8f syn = {}, mem = {}, mem2max;
#pragma unroll
  for (int r = 0; r < 8; ++r) mem2max[r] = -3.4e38f;

  const int mbase = half_hi * 8;                 // D row = r + mbase
  const int koff  = half_hi ? 8 : 0;
  const float* curp = cur_ff + CUR_IDX(0, btile, ncol, mbase);

  __syncthreads();                               // zeroed spkLDS visible

  for (int t = 0; t < T_SZ; ++t) {
    v8f spk;
    if (is_l1) {
      v8f acc = {};                              // rec = spk1(t-1) @ Wfb^T tile
#pragma unroll
      for (int c = 0; c < 7; ++c) {
        const half_t* base = &spkLDS[mn * NHP + c * 32 + koff];
        v8h lo = *(const v8h*)(base);
        v8h hi = *(const v8h*)(base + 16);
        v16h a = __builtin_shufflevector(lo, hi, 0,1,2,3,4,5,6,7,
                                                 8,9,10,11,12,13,14,15);
        acc = __builtin_amdgcn_wmma_f32_16x16x32_f16(false, a, false, bfrag[c],
                                                     (short)0, acc, false, false);
      }
      // contiguous 8-float feed-forward current (2x global_load_b128)
      v4f c0 = *(const v4f*)(curp);
      v4f c1 = *(const v4f*)(curp + 4);
      if (t + 1 < T_SZ) __builtin_prefetch(curp + TSTRIDE, 0, 1);
      curp += TSTRIDE;
      // snntorch Synaptic update (reset-by-subtract from previous mem)
#pragma unroll
      for (int r = 0; r < 8; ++r) {
        float cff   = (r < 4) ? c0[r & 3] : c1[r & 3];
        float cur   = cff + acc[r] + p_bias;
        float reset = (mem[r] > p_thr) ? p_thr : 0.0f;
        syn[r] = p_alpha * syn[r] + cur;
        mem[r] = p_beta  * mem[r] + syn[r] - reset;
        spk[r] = (mem[r] > p_thr) ? 1.0f : 0.0f;
      }
    }
    __syncthreads();                             // old spikes fully consumed
    if (is_l1) {
#pragma unroll
      for (int r = 0; r < 8; ++r)
        spkLDS[(r + mbase) * NHP + ncol] = (half_t)spk[r];
    }
    __syncthreads();                             // spk1(t) visible
    if (!is_l1) {
      v8f acc = {};                              // cur2 = spk1(t) @ W2^T
#pragma unroll
      for (int c = 0; c < 7; ++c) {
        const half_t* base = &spkLDS[mn * NHP + c * 32 + koff];
        v8h lo = *(const v8h*)(base);
        v8h hi = *(const v8h*)(base + 16);
        v16h a = __builtin_shufflevector(lo, hi, 0,1,2,3,4,5,6,7,
                                                 8,9,10,11,12,13,14,15);
        acc = __builtin_amdgcn_wmma_f32_16x16x32_f16(false, a, false, bfrag[c],
                                                     (short)0, acc, false, false);
      }
#pragma unroll
      for (int r = 0; r < 8; ++r) {
        float cur   = acc[r] + p_bias;
        float reset = (mem[r] > p_thr) ? p_thr : 0.0f;
        syn[r] = p_alpha * syn[r] + cur;
        mem[r] = p_beta  * mem[r] + syn[r] - reset;
        mem2max[r] = fmaxf(mem2max[r], mem[r]);
      }
    }
  }

  if (!is_l1) {
#pragma unroll
    for (int r = 0; r < 8; ++r)
      smax[(r + mbase) * 32 + ncol] = mem2max[r];
  }
  __syncthreads();
  if (tid < 16) {
    const int m = tid;
    float mx = -3.4e38f;
    for (int o = 0; o < NOUT; ++o) mx = fmaxf(mx, smax[m * 32 + o]);
    float s = 0.0f;
    for (int o = 0; o < NOUT; ++o) s += __expf(smax[m * 32 + o] - mx);
    float inv = 1.0f / s;
    for (int o = 0; o < NOUT; ++o)
      out[(b0 + m) * NOUT + o] = __expf(smax[m * 32 + o] - mx) * inv;
  }
}

// ---------------------------------------------------------------------------
extern "C" void kernel_launch(void* const* d_in, const int* in_sizes, int n_in,
                              void* d_out, int out_size, void* d_ws, size_t ws_size,
                              hipStream_t stream) {
  const float* x      = (const float*)d_in[0];   // [B,T,1,NIN]
  const float* W1     = (const float*)d_in[1];   // [NH,NIN]
  const float* b1     = (const float*)d_in[2];
  const float* Wfb    = (const float*)d_in[3];   // [NH,NH]
  const float* bfb    = (const float*)d_in[4];
  const float* W2     = (const float*)d_in[5];   // [NOUT,NH]
  const float* b2     = (const float*)d_in[6];
  const float* alpha1 = (const float*)d_in[7];
  const float* beta1  = (const float*)d_in[8];
  const float* thr1   = (const float*)d_in[9];
  const float* alpha2 = (const float*)d_in[10];
  const float* beta2  = (const float*)d_in[11];
  const float* thr2   = (const float*)d_in[12];
  float* out = (float*)d_out;

  // Workspace layout (all offsets 16B-aligned):
  char* ws = (char*)d_ws;
  float*  cur_ff = (float*)ws;                         // 500*8*224*16*4 = 57,344,000 B
  half_t* xh     = (half_t*)(ws + 57344000);           // 64000*704*2    = 90,112,000 B
  half_t* W1h    = (half_t*)(ws + 57344000 + 90112000);            // 224*704*2
  half_t* Wfbh   = (half_t*)(ws + 57344000 + 90112000 + 315392);   // 224*224*2
  half_t* W2h    = (half_t*)(ws + 57344000 + 90112000 + 315392 + 100352); // 32*224*2

  // Phase 0: zero-padded f16 repacks (branch-free WMMA loads downstream)
  pack_f16<<<2048, 256, 0, stream>>>(x,   xh,   MROWS, NIN, MROWS, NINP);
  pack_f16<<<64,   256, 0, stream>>>(W1,  W1h,  NH,    NIN, NHP,   NINP);
  pack_f16<<<32,   256, 0, stream>>>(Wfb, Wfbh, NH,    NH,  NHP,   NHP);
  pack_f16<<<8,    256, 0, stream>>>(W2,  W2h,  NOUT,  NH,  32,    NHP);

  // Phase 1: feed-forward GEMM, 4096 padded M-tiles * 14 N-tiles
  snn_ff_gemm<<<TILES / 8, 256, 0, stream>>>(xh, W1h, b1, cur_ff);

  // Phase 2: 8 independent batch shards run the full T=500 scan
  snn_scan<<<8, 512, 0, stream>>>(cur_ff, Wfbh, W2h, bfb, b2,
                                  alpha1, beta1, thr1, alpha2, beta2, thr2, out);
}
